// WeightedHGTConv_8375186227282
// MI455X (gfx1250) — matrix operand order repaired and checked by reading, not compiled, not run on hardware
//
#include <hip/hip_runtime.h>

#define N_NODES 50000
#define N_EDGES 400000
#define DIM 128
#define NTY 4
#define NR 8
#define NH 8
#define DKH 16
#define TILES_PER_BLOCK 5   // 3125 node tiles = 625 blocks x 5

typedef __attribute__((ext_vector_type(16))) __bf16 v16bf;
typedef __attribute__((ext_vector_type(8)))  float  v8f;

// ---- sortable-uint encoding for float atomic max over any sign ----
__device__ __forceinline__ unsigned enc_f32(float f) {
    unsigned b = __float_as_uint(f);
    return (b & 0x80000000u) ? ~b : (b | 0x80000000u);
}
__device__ __forceinline__ float dec_f32(unsigned u) {
    return (u & 0x80000000u) ? __uint_as_float(u & 0x7FFFFFFFu)
                             : __uint_as_float(~u);
}

// ---------------- init scratch (accum, ssum, smax) ----------------
__global__ void init_ws_kernel(float* __restrict__ accum,
                               float* __restrict__ ssum,
                               unsigned* __restrict__ smax) {
    int idx = blockIdx.x * blockDim.x + threadIdx.x;
    if (idx < N_NODES * DIM) accum[idx] = 0.0f;
    if (idx < N_NODES * NH) { ssum[idx] = 0.0f; smax[idx] = 0u; }
}

// -------- pack Wq/Wk/Wv [T,128,128] f32 -> bf16 WMMA B-lane layout --------
// element index: ((((m*4+t)*8+nt)*4+kc)*32+lane)*16 + j
// For fixed (m,nt): chunks c = t*4+kc form the augmented-K (512) stack of the
// four per-type weight matrices.
// B packing used here (contiguous k, matches A): lanes 0-15 -> col N=lane, K=kc*32+0..15
//                           lanes 16-31 -> col N=lane-16, K=kc*32+16..31
__global__ void pack_weights_kernel(const float* __restrict__ Wq,
                                    const float* __restrict__ Wk,
                                    const float* __restrict__ Wv,
                                    __bf16* __restrict__ wpk) {
    int idx = blockIdx.x * blockDim.x + threadIdx.x;
    if (idx >= 3 * NTY * 8 * 4 * 32 * 16) return;
    int j    = idx & 15;
    int lane = (idx >> 4) & 31;
    int kc   = (idx >> 9) & 3;
    int nt   = (idx >> 11) & 7;
    int t    = (idx >> 14) & 3;
    int m    = idx >> 16;
    int k = kc * 32 + ((lane & 16) ? 16 : 0) + j;
    int n = nt * 16 + (lane & 15);
    const float* W = (m == 0) ? Wq : ((m == 1) ? Wk : Wv);
    wpk[idx] = (__bf16)W[(t * DIM + k) * DIM + n];
}

// ---------------- QKV projection via bf16 WMMA, K-concatenated types -------
// The per-row type selection  out = sum_t (mask_t . A) @ B_t  is computed as
// ONE GEMM over an augmented K=512: B_aug stacks the 4 per-type weights
// (still 16 v16bf = 128 VGPRs, register-stationary per wave); A_aug places
// each row's data in its own type's 128-wide K-block and zeros elsewhere.
// Inner loop is therefore pure ds_load_b128 + v_wmma with a clean D->C
// accumulation chain: no per-WMMA cndmask masking, no D->A/B hazard NOPs.
// gridDim = (625, 3): blockIdx.y = matrix m; 8 waves = 8 n-tiles; 5 node
// tiles per block with the 16KB A_aug tile staged through LDS.
__global__ void __launch_bounds__(256)
qkv_gemm_kernel(const float* __restrict__ x, const int* __restrict__ ntype,
                const __bf16* __restrict__ wpk,
                const float* __restrict__ bq, const float* __restrict__ bk,
                const float* __restrict__ bv,
                float* __restrict__ Q, float* __restrict__ K,
                float* __restrict__ V) {
    __shared__ __align__(32) __bf16 xpk[16][32][16];  // [t*4+kc][lane][j] 16KB
    __shared__ int tTypes[16];

    const int tid  = threadIdx.x;
    const int lane = tid & 31;
    const int nt   = tid >> 5;            // wave id == n-tile
    const int m    = blockIdx.y;          // 0=Q 1=K 2=V

    const float* bias = (m == 0) ? bq : ((m == 1) ? bk : bv);
    float* out        = (m == 0) ? Q : ((m == 1) ? K : V);

    // B_aug stationary in VGPRs: chunk c = t*4+kc
    v16bf B[16];
#pragma unroll
    for (int c = 0; c < 16; c++)
        B[c] = ((const v16bf*)wpk)[((((m * 4 + (c >> 2)) * 8 + nt) * 4 + (c & 3)) * 32 + lane];

    const int srow = tid >> 4;            // staging row (one per thread)

    for (int it = 0; it < TILES_PER_BLOCK; it++) {
        int tile = blockIdx.x * TILES_PER_BLOCK + it;
        __syncthreads();                   // previous tile fully consumed

        // zero-fill A_aug (16KB) + load row types for epilogue
        {
            uint4  z4 = {0u, 0u, 0u, 0u};
            uint4* zp = (uint4*)&xpk[0][0][0];
#pragma unroll
            for (int q = 0; q < 4; q++) zp[q * 256 + tid] = z4;
            if (tid < 16) tTypes[tid] = ntype[tile * 16 + tid];
        }
        __syncthreads();

        // stage this thread's row segment into its type's K-block,
        // contiguous-k A-lane order (same as B) within each 32-wide chunk:
        // lanes 0-15: row=lane, K = 0..15; lanes 16-31: K = 16..31
        const int myT = ntype[tile * 16 + srow];
#pragma unroll
        for (int q = 0; q < 8; q++) {
            int k  = (tid & 15) * 8 + q;
            int kr = k & 31;
            int c  = myT * 4 + (k >> 5);
            int l2 = srow + ((kr & 16) ? 16 : 0);
            int j  = kr & 15;
            xpk[c][l2][j] = (__bf16)x[(tile * 16 + srow) * DIM + k];
        }
        // prefetch next tile's rows into cache while we do WMMAs
        if (it + 1 < TILES_PER_BLOCK)
            __builtin_prefetch(&x[((tile + 1) * 16 + srow) * DIM + (tid & 15) * 8], 0, 1);
        __syncthreads();

        v8f acc = {};
#pragma unroll
        for (int c = 0; c < 16; c++) {
            v16bf a = *(const v16bf*)&xpk[c][lane][0];
            acc = __builtin_amdgcn_wmma_f32_16x16x32_bf16(
                false, a, false, B[c], (short)0, acc, false, false);
        }
        int col = nt * 16 + (lane & 15);
#pragma unroll
        for (int r = 0; r < 8; r++) {
            int rowLocal = r + ((lane & 16) ? 8 : 0);   // C layout
            out[(tile * 16 + rowLocal) * DIM + col] =
                acc[r] + bias[tTypes[rowLocal] * DIM + col];
        }
    }
}

// ---------------- edge scores + segment max ----------------
__global__ void edge_scores_kernel(const float* __restrict__ Q,
                                   const float* __restrict__ K,
                                   const int* __restrict__ ei,
                                   const int* __restrict__ etype,
                                   const int* __restrict__ esign,
                                   const float* __restrict__ rel_q,
                                   const float* __restrict__ rel_k,
                                   const float* __restrict__ sk_neutral,
                                   const float* __restrict__ rel_bias,
                                   float* __restrict__ scores,
                                   unsigned* __restrict__ smax) {
    int idx = blockIdx.x * blockDim.x + threadIdx.x;
    if (idx >= N_EDGES * NH) return;
    int e = idx >> 3, h = idx & 7;
    int src = ei[e], dst = ei[N_EDGES + e];
    int r = etype[e], s = esign[e];
    const float* qp = &Q[dst * DIM + h * DKH];
    const float* kp = &K[src * DIM + h * DKH];
    const float* rq = &rel_q[(r * NH + h) * DKH];
    const float* rk = &rel_k[(r * NH + h) * DKH];
    float acc = 0.0f;
#pragma unroll
    for (int d = 0; d < DKH; d++) {
        float sk = (s == 0) ? sk_neutral[h * DKH + d] : (float)s;
        acc += qp[d] * rq[d] * kp[d] * rk[d] * sk;
    }
    float sc = acc * 0.25f + rel_bias[r * NH + h];  // 1/sqrt(16)
    scores[idx] = sc;
    atomicMax(&smax[dst * NH + h], enc_f32(sc));
}

// ---------------- exp + segment sum ----------------
__global__ void edge_exp_kernel(const int* __restrict__ ei,
                                const unsigned* __restrict__ smax,
                                float* __restrict__ scores,
                                float* __restrict__ ssum) {
    int idx = blockIdx.x * blockDim.x + threadIdx.x;
    if (idx >= N_EDGES * NH) return;
    int e = idx >> 3, h = idx & 7;
    int dst = ei[N_EDGES + e];
    float m  = dec_f32(smax[dst * NH + h]);
    float ex = __expf(scores[idx] - m);
    scores[idx] = ex;
    atomicAdd(&ssum[dst * NH + h], ex);
}

// ---------------- normalized weighted scatter ----------------
__global__ void edge_scatter_kernel(const float* __restrict__ V,
                                    const int* __restrict__ ei,
                                    const int* __restrict__ etype,
                                    const int* __restrict__ esign,
                                    const float* __restrict__ rel_v,
                                    const float* __restrict__ sv_neutral,
                                    const float* __restrict__ scores,
                                    const float* __restrict__ ssum,
                                    float* __restrict__ accum) {
    int idx = blockIdx.x * blockDim.x + threadIdx.x;
    if (idx >= N_EDGES * NH) return;
    int e = idx >> 3, h = idx & 7;
    int src = ei[e], dst = ei[N_EDGES + e];
    int r = etype[e], s = esign[e];
    float attn = scores[idx] / (ssum[dst * NH + h] + 1e-16f);
    const float* vp = &V[src * DIM + h * DKH];
    const float* rv = &rel_v[(r * NH + h) * DKH];
#pragma unroll
    for (int d = 0; d < DKH; d++) {
        float sv = (s == 0) ? sv_neutral[h * DKH + d] : (float)s;
        atomicAdd(&accum[dst * DIM + h * DKH + d], vp[d] * rv[d] * sv * attn);
    }
}

// ---------------- gated skip + LayerNorm (one wave per node) ----------------
__global__ void __launch_bounds__(256)
finalize_kernel(const float* __restrict__ x, const int* __restrict__ ntype,
                const float* __restrict__ accum, const float* __restrict__ skip,
                const float* __restrict__ gamma, const float* __restrict__ beta,
                float* __restrict__ out) {
    int lane = threadIdx.x & 31;
    int wave = threadIdx.x >> 5;
    int node = blockIdx.x * 8 + wave;
    if (node >= N_NODES) return;
    int t = ntype[node];
    float alpha = 1.0f / (1.0f + __expf(-skip[t]));
    float pre[4];
    float s = 0.0f, s2 = 0.0f;
#pragma unroll
    for (int i = 0; i < 4; i++) {
        int d = i * 32 + lane;
        float p = alpha * accum[node * DIM + d] + (1.0f - alpha) * x[node * DIM + d];
        pre[i] = p;
        s += p; s2 += p * p;
    }
#pragma unroll
    for (int off = 16; off > 0; off >>= 1) {
        s  += __shfl_xor(s, off, 32);
        s2 += __shfl_xor(s2, off, 32);
    }
    float mu   = s * (1.0f / DIM);
    float var  = s2 * (1.0f / DIM) - mu * mu;
    float rstd = rsqrtf(var + 1e-5f);
#pragma unroll
    for (int i = 0; i < 4; i++) {
        int d = i * 32 + lane;
        out[node * DIM + d] = (pre[i] - mu) * rstd * gamma[t * DIM + d] + beta[t * DIM + d];
    }
}

extern "C" void kernel_launch(void* const* d_in, const int* in_sizes, int n_in,
                              void* d_out, int out_size, void* d_ws, size_t ws_size,
                              hipStream_t stream) {
    const float* node_inp  = (const float*)d_in[0];
    const int*   node_type = (const int*)d_in[1];
    const int*   edge_index= (const int*)d_in[2];
    const int*   edge_type = (const int*)d_in[3];
    const int*   edge_sign = (const int*)d_in[4];
    const float* Wq = (const float*)d_in[5];  const float* bq = (const float*)d_in[6];
    const float* Wk = (const float*)d_in[7];  const float* bk = (const float*)d_in[8];
    const float* Wv = (const float*)d_in[9];  const float* bv = (const float*)d_in[10];
    const float* rel_q = (const float*)d_in[11];
    const float* rel_k = (const float*)d_in[12];
    const float* rel_v = (const float*)d_in[13];
    const float* skn   = (const float*)d_in[14];
    const float* svn   = (const float*)d_in[15];
    const float* rbias = (const float*)d_in[16];
    const float* skip  = (const float*)d_in[17];
    const float* gamma = (const float*)d_in[18];
    const float* beta  = (const float*)d_in[19];

    char* ws = (char*)d_ws;
    size_t off = 0;
    __bf16*   wpk    = (__bf16*)(ws + off);   off += (size_t)3 * NTY * DIM * DIM * 2;       // 393216
    float*    Q      = (float*)(ws + off);    off += (size_t)N_NODES * DIM * 4;             // 25.6MB
    float*    K      = (float*)(ws + off);    off += (size_t)N_NODES * DIM * 4;
    float*    V      = (float*)(ws + off);    off += (size_t)N_NODES * DIM * 4;
    float*    scores = (float*)(ws + off);    off += (size_t)N_EDGES * NH * 4;              // 12.8MB
    unsigned* smax   = (unsigned*)(ws + off); off += (size_t)N_NODES * NH * 4;
    float*    ssum   = (float*)(ws + off);    off += (size_t)N_NODES * NH * 4;
    float*    accum  = (float*)(ws + off);    off += (size_t)N_NODES * DIM * 4;
    float*    out    = (float*)d_out;

    init_ws_kernel<<<(N_NODES * DIM + 255) / 256, 256, 0, stream>>>(accum, ssum, smax);
    pack_weights_kernel<<<(3 * NTY * DIM * DIM + 255) / 256, 256, 0, stream>>>(Wq, Wk, Wv, wpk);
    qkv_gemm_kernel<<<dim3(N_NODES / 16 / TILES_PER_BLOCK, 3), 256, 0, stream>>>(
        node_inp, node_type, wpk, bq, bk, bv, Q, K, V);
    int eblk = (N_EDGES * NH + 255) / 256;
    edge_scores_kernel<<<eblk, 256, 0, stream>>>(Q, K, edge_index, edge_type, edge_sign,
                                                 rel_q, rel_k, skn, rbias, scores, smax);
    edge_exp_kernel<<<eblk, 256, 0, stream>>>(edge_index, smax, scores, ssum);
    edge_scatter_kernel<<<eblk, 256, 0, stream>>>(V, edge_index, edge_type, edge_sign,
                                                  rel_v, svn, scores, ssum, accum);
    finalize_kernel<<<(N_NODES + 7) / 8, 256, 0, stream>>>(node_inp, node_type, accum,
                                                           skip, gamma, beta, out);
}
